// SmilesRNNVAE_28398323761526
// MI455X (gfx1250) — compile-verified, matrix operand order checked by reading, not run on hardware
//
#include <hip/hip_runtime.h>

// SmilesRNNVAE for MI455X (gfx1250): f16 WMMA (16x16x32, f32 accum) GEMMs,
// persistent per-batch-tile recurrent kernels, gates staged in LDS.
// Weights pre-swizzled into WMMA B-fragment order so every fragment load is a
// contiguous 32B vector load per lane; A fragments load as 2x16B from LDS.
// Wave indices go through readfirstlane so all WMMA loops are scalar-controlled
// (EXEC stays all-ones, per ISA 7.12 requirement).

#define BATCH 4096
#define TLEN  128
#define KN    64    // vocab
#define ED    35    // embed dim
#define LD    200   // latent dim

typedef __attribute__((ext_vector_type(16))) _Float16 v16h;
typedef __attribute__((ext_vector_type(8)))  _Float16 v8h;
typedef __attribute__((ext_vector_type(8)))  float    v8f;

// ---------------- WMMA fragment helpers (ISA 7.12.2 layouts) ----------------
// A (16x32 f16): lane l holds row M=l&15; halves i map to
// K = (i&7) + ((i>>3)<<4) + ((lane>=16)?8:0) -> two contiguous 8-half runs.
__device__ __forceinline__ v16h load_frag_a(const _Float16* A, int lda) {
  const int lane = threadIdx.x & 31;
  const int m    = lane & 15;
  const int kh   = (lane >> 4) << 3;
  const _Float16* p = A + m * lda + kh;
  v8h lo = *(const v8h*)(p);        // K = kh .. kh+7
  v8h hi = *(const v8h*)(p + 16);   // K = kh+16 .. kh+23
  return __builtin_shufflevector(lo, hi, 0, 1, 2, 3, 4, 5, 6, 7,
                                 8, 9, 10, 11, 12, 13, 14, 15);
}

// B tiles pre-swizzled to [tile][lane][16 halves]: one 32B load per lane.
__device__ __forceinline__ v16h load_frag_b_sw(const _Float16* Bt) {
  const int lane = threadIdx.x & 31;
  return ((const v16h*)Bt)[lane];
}

__device__ __forceinline__ const _Float16* frag_ptr(const _Float16* B, int nt,
                                                    int kt, int nkt) {
  return B + (((size_t)nt * nkt + kt) << 9);  // 512 halves per 32x16 tile
}

// C/D (16x16 f32): VGPR v -> M = v + (lane>=16 ? 8 : 0), N = lane&15.
__device__ __forceinline__ void store_frag_c(float* C, int ldc, v8f acc) {
  const int lane = threadIdx.x & 31;
  const int n    = lane & 15;
  const int mb   = (lane >> 4) << 3;
#pragma unroll
  for (int v = 0; v < 8; ++v) C[(mb + v) * ldc + n] = acc[v];
}

__device__ __forceinline__ v8f wmma16(v16h a, v16h b, v8f c) {
  return __builtin_amdgcn_wmma_f32_16x16x32_f16(false, a, false, b, (short)0, c,
                                                false, false);
}

__device__ __forceinline__ float sigm(float x) { return 1.f / (1.f + __expf(-x)); }

// Wave id as a compiler-known scalar -> WMMA tile loops use SALU control flow.
__device__ __forceinline__ int wave_id() {
  return __builtin_amdgcn_readfirstlane(threadIdx.x >> 5);
}

// --------- weight prep: f32 [N,K] -> f16 swizzled B fragments ---------------
// out layout: [nt][kt][lane 0..31][i 0..15], tile = 32x16 of B^T (K-major).
__global__ void convert_wT_swz_kernel(const float* __restrict__ W,
                                      _Float16* __restrict__ out,
                                      int Nrows, int Kdim, int Kpad, int Npad) {
  int idx = blockIdx.x * blockDim.x + threadIdx.x;
  if (idx >= Kpad * Npad) return;
  int i    = idx & 15;
  int lane = (idx >> 4) & 31;
  int tile = idx >> 9;
  int nkt  = Kpad >> 5;
  int nt = tile / nkt, kt = tile % nkt;
  int n = nt * 16 + (lane & 15);
  int k = kt * 32 + ((lane >> 4) << 3) + (i & 7) + ((i >> 3) << 4);
  out[idx] = (k < Kdim && n < Nrows) ? (_Float16)W[n * Kdim + k] : (_Float16)0.f;
}

// ---------------------------- encoder (persistent) --------------------------
__global__ __launch_bounds__(256) void encoder_kernel(
    const int* __restrict__ x, const float* __restrict__ emb,
    const _Float16* __restrict__ Bih0, const _Float16* __restrict__ Bhh0,
    const _Float16* __restrict__ Bih1, const _Float16* __restrict__ Bhh1,
    const float* __restrict__ bih0, const float* __restrict__ bhh0,
    const float* __restrict__ bih1, const float* __restrict__ bhh1,
    _Float16* __restrict__ h2out /* [B,224] */) {
  __shared__ __align__(32) _Float16 xa[16 * 64];   // step input, pad 35->64
  __shared__ __align__(32) _Float16 h1[16 * 224];  // layer0 hidden, pad 224
  __shared__ __align__(32) _Float16 h2[16 * 224];  // layer1 hidden
  __shared__ __align__(32) float    gi[16 * 608];  // input gates (600 valid)
  __shared__ __align__(32) float    gh[16 * 608];  // hidden gates

  const int b0 = blockIdx.x * 16;
  const int tid = threadIdx.x;
  const int wave = wave_id();

  if (tid < 32) {  // warm weight images toward this WGP (global_prefetch_b8)
    __builtin_prefetch(Bih0 + (size_t)tid * 1216, 0, 2);
    __builtin_prefetch(Bhh0 + (size_t)tid * 4256, 0, 2);
    __builtin_prefetch(Bih1 + (size_t)tid * 4256, 0, 2);
    __builtin_prefetch(Bhh1 + (size_t)tid * 4256, 0, 2);
  }
  for (int i = tid; i < 16 * 224; i += 256) { h1[i] = (_Float16)0.f; h2[i] = (_Float16)0.f; }
  __syncthreads();

  for (int t = 0; t < TLEN; ++t) {
    for (int i = tid; i < 16 * 64; i += 256) {
      int m = i >> 6, k = i & 63;
      int idx = x[(size_t)(b0 + m) * TLEN + t];
      xa[i] = (k < ED) ? (_Float16)emb[idx * ED + k] : (_Float16)0.f;
    }
    __syncthreads();

    // layer0: gi = x@Wih^T (K=64), gh = h1@Whh^T (K=224); N=608 (38 tiles)
    {
      v16h ax[2], ah[7];
#pragma unroll
      for (int kt = 0; kt < 2; ++kt) ax[kt] = load_frag_a(xa + kt * 32, 64);
#pragma unroll
      for (int kt = 0; kt < 7; ++kt) ah[kt] = load_frag_a(h1 + kt * 32, 224);
      for (int nt = wave; nt < 38; nt += 8) {
        v8f a0 = {};
#pragma unroll
        for (int kt = 0; kt < 2; ++kt)
          a0 = wmma16(ax[kt], load_frag_b_sw(frag_ptr(Bih0, nt, kt, 2)), a0);
        store_frag_c(gi + nt * 16, 608, a0);
        v8f a1 = {};
#pragma unroll
        for (int kt = 0; kt < 7; ++kt)
          a1 = wmma16(ah[kt], load_frag_b_sw(frag_ptr(Bhh0, nt, kt, 7)), a1);
        store_frag_c(gh + nt * 16, 608, a1);
      }
    }
    __syncthreads();

    for (int i = tid; i < 16 * LD; i += 256) {
      int m = i / LD, j = i % LD;
      float ir = gi[m * 608 + j]           + bih0[j];
      float iz = gi[m * 608 + LD + j]      + bih0[LD + j];
      float in_ = gi[m * 608 + 2 * LD + j] + bih0[2 * LD + j];
      float hr = gh[m * 608 + j]           + bhh0[j];
      float hz = gh[m * 608 + LD + j]      + bhh0[LD + j];
      float hn = gh[m * 608 + 2 * LD + j]  + bhh0[2 * LD + j];
      float r = sigm(ir + hr), zg = sigm(iz + hz);
      float n = tanhf(in_ + r * hn);
      float hp = (float)h1[m * 224 + j];
      h1[m * 224 + j] = (_Float16)((1.f - zg) * n + zg * hp);
    }
    __syncthreads();

    // layer1: gi = h1@Wih1^T, gh = h2@Whh1^T (both K=224)
    {
      v16h a1f[7], a2f[7];
#pragma unroll
      for (int kt = 0; kt < 7; ++kt) {
        a1f[kt] = load_frag_a(h1 + kt * 32, 224);
        a2f[kt] = load_frag_a(h2 + kt * 32, 224);
      }
      for (int nt = wave; nt < 38; nt += 8) {
        v8f a0 = {}, a1 = {};
#pragma unroll
        for (int kt = 0; kt < 7; ++kt) {
          a0 = wmma16(a1f[kt], load_frag_b_sw(frag_ptr(Bih1, nt, kt, 7)), a0);
          a1 = wmma16(a2f[kt], load_frag_b_sw(frag_ptr(Bhh1, nt, kt, 7)), a1);
        }
        store_frag_c(gi + nt * 16, 608, a0);
        store_frag_c(gh + nt * 16, 608, a1);
      }
    }
    __syncthreads();

    for (int i = tid; i < 16 * LD; i += 256) {
      int m = i / LD, j = i % LD;
      float ir = gi[m * 608 + j]           + bih1[j];
      float iz = gi[m * 608 + LD + j]      + bih1[LD + j];
      float in_ = gi[m * 608 + 2 * LD + j] + bih1[2 * LD + j];
      float hr = gh[m * 608 + j]           + bhh1[j];
      float hz = gh[m * 608 + LD + j]      + bhh1[LD + j];
      float hn = gh[m * 608 + 2 * LD + j]  + bhh1[2 * LD + j];
      float r = sigm(ir + hr), zg = sigm(iz + hz);
      float n = tanhf(in_ + r * hn);
      float hp = (float)h2[m * 224 + j];
      h2[m * 224 + j] = (_Float16)((1.f - zg) * n + zg * hp);
    }
    __syncthreads();
  }

  for (int i = tid; i < 16 * 224; i += 256)
    h2out[(size_t)(b0 + (i / 224)) * 224 + (i % 224)] = h2[i];
}

// ---------------- latent: mu / logvar / z / predY (WMMA) --------------------
__global__ __launch_bounds__(256) void latent_kernel(
    const _Float16* __restrict__ h2f16, const _Float16* __restrict__ fc11t,
    const _Float16* __restrict__ fc12t, const _Float16* __restrict__ p1t,
    const float* __restrict__ fc11_b, const float* __restrict__ fc12_b,
    const float* __restrict__ p1_b, const float* __restrict__ p2_w,
    const float* __restrict__ p2_b, const float* __restrict__ eps,
    float* __restrict__ out_mu, float* __restrict__ out_lv,
    float* __restrict__ out_py, _Float16* __restrict__ zf16) {
  __shared__ __align__(32) float    smu[16 * 208];
  __shared__ __align__(32) float    slv[16 * 208];
  __shared__ __align__(32) _Float16 muh[16 * 224];

  const int b0 = blockIdx.x * 16;
  const int tid = threadIdx.x;
  const int wave = wave_id();
  const _Float16* A = h2f16 + (size_t)b0 * 224;

  {
    v16h af[7];
#pragma unroll
    for (int kt = 0; kt < 7; ++kt) af[kt] = load_frag_a(A + kt * 32, 224);
    for (int nt = wave; nt < 13; nt += 8) {
      v8f a1 = {}, a2 = {};
#pragma unroll
      for (int kt = 0; kt < 7; ++kt) {
        a1 = wmma16(af[kt], load_frag_b_sw(frag_ptr(fc11t, nt, kt, 7)), a1);
        a2 = wmma16(af[kt], load_frag_b_sw(frag_ptr(fc12t, nt, kt, 7)), a2);
      }
      store_frag_c(smu + nt * 16, 208, a1);
      store_frag_c(slv + nt * 16, 208, a2);
    }
  }
  for (int i = tid; i < 16 * 224; i += 256) muh[i] = (_Float16)0.f;
  __syncthreads();

  for (int i = tid; i < 16 * LD; i += 256) {
    int m = i / LD, j = i % LD;
    int b = b0 + m;
    float mu = smu[m * 208 + j] + fc11_b[j];
    float lv = slv[m * 208 + j] + fc12_b[j];
    out_mu[(size_t)b * LD + j] = mu;
    out_lv[(size_t)b * LD + j] = lv;
    float zv = mu + eps[(size_t)b * LD + j] * __expf(0.5f * lv);
    zf16[(size_t)b * LD + j] = (_Float16)zv;
    muh[m * 224 + j] = (_Float16)mu;
  }
  __syncthreads();

  {
    v16h af[7];
#pragma unroll
    for (int kt = 0; kt < 7; ++kt) af[kt] = load_frag_a(muh + kt * 32, 224);
    for (int nt = wave; nt < 13; nt += 8) {
      v8f a = {};
#pragma unroll
      for (int kt = 0; kt < 7; ++kt)
        a = wmma16(af[kt], load_frag_b_sw(frag_ptr(p1t, nt, kt, 7)), a);
      store_frag_c(smu + nt * 16, 208, a);
    }
  }
  __syncthreads();

  if (tid < 16) {
    int m = tid;
    float s = p2_b[0];
    for (int j = 0; j < LD; ++j) {
      float v = smu[m * 208 + j] + p1_b[j];
      s += (v > 0.f ? v : 0.f) * p2_w[j];
    }
    out_py[b0 + m] = s;
  }
}

// ---------------------------- decoder (persistent) --------------------------
// Zero initial hidden each step => gh == bhh, h = (1-z)*n. fc2 fused.
__global__ __launch_bounds__(128) void decoder_kernel(
    const _Float16* __restrict__ zf16, const _Float16* __restrict__ dih0t,
    const _Float16* __restrict__ dih1t, const _Float16* __restrict__ fc2t,
    const float* __restrict__ dbih0, const float* __restrict__ dbhh0,
    const float* __restrict__ dbih1, const float* __restrict__ dbhh1,
    const float* __restrict__ fc2_b, float* __restrict__ pred) {
  __shared__ __align__(32) _Float16 xin[16 * 256];  // [z(200)|lastWord(35)|0]
  __shared__ __align__(32) _Float16 h1h[16 * 64];
  __shared__ __align__(32) _Float16 h2h[16 * 64];
  __shared__ __align__(32) float    g[16 * 112];    // 105 valid gate columns

  const int b0 = blockIdx.x * 16;
  const int tid = threadIdx.x;
  const int wave = wave_id();

  if (tid < 32) {
    __builtin_prefetch(dih0t + (size_t)tid * 896, 0, 2);
    __builtin_prefetch(dih1t + (size_t)tid * 224, 0, 2);
  }
  for (int i = tid; i < 16 * 256; i += 128) {
    int m = i >> 8, j = i & 255;
    xin[i] = (j < LD) ? zf16[(size_t)(b0 + m) * LD + j] : (_Float16)0.f;
  }
  for (int i = tid; i < 16 * 64; i += 128) { h1h[i] = (_Float16)0.f; h2h[i] = (_Float16)0.f; }
  __syncthreads();

  for (int t = 0; t < TLEN; ++t) {
    // layer0: gi = xin(16x256) @ dW_ih0^T  (N=112 -> 7 tiles, K=256 -> 8)
    {
      v16h af[8];
#pragma unroll
      for (int kt = 0; kt < 8; ++kt) af[kt] = load_frag_a(xin + kt * 32, 256);
      for (int nt = wave; nt < 7; nt += 4) {
        v8f a = {};
#pragma unroll
        for (int kt = 0; kt < 8; ++kt)
          a = wmma16(af[kt], load_frag_b_sw(frag_ptr(dih0t, nt, kt, 8)), a);
        store_frag_c(g + nt * 16, 112, a);
      }
    }
    __syncthreads();

    for (int i = tid; i < 16 * ED; i += 128) {
      int m = i / ED, j = i % ED;
      float ir = g[m * 112 + j]           + dbih0[j];
      float iz = g[m * 112 + ED + j]      + dbih0[ED + j];
      float in_ = g[m * 112 + 2 * ED + j] + dbih0[2 * ED + j];
      float r = sigm(ir + dbhh0[j]);
      float zg = sigm(iz + dbhh0[ED + j]);
      float n = tanhf(in_ + r * dbhh0[2 * ED + j]);
      h1h[m * 64 + j] = (_Float16)((1.f - zg) * n);
    }
    __syncthreads();

    // layer1: gi = h1(16x64) @ dW_ih1^T
    {
      v16h af[2];
#pragma unroll
      for (int kt = 0; kt < 2; ++kt) af[kt] = load_frag_a(h1h + kt * 32, 64);
      for (int nt = wave; nt < 7; nt += 4) {
        v8f a = {};
#pragma unroll
        for (int kt = 0; kt < 2; ++kt)
          a = wmma16(af[kt], load_frag_b_sw(frag_ptr(dih1t, nt, kt, 2)), a);
        store_frag_c(g + nt * 16, 112, a);
      }
    }
    __syncthreads();

    for (int i = tid; i < 16 * ED; i += 128) {
      int m = i / ED, j = i % ED;
      float ir = g[m * 112 + j]           + dbih1[j];
      float iz = g[m * 112 + ED + j]      + dbih1[ED + j];
      float in_ = g[m * 112 + 2 * ED + j] + dbih1[2 * ED + j];
      float r = sigm(ir + dbhh1[j]);
      float zg = sigm(iz + dbhh1[ED + j]);
      float n = tanhf(in_ + r * dbhh1[2 * ED + j]);
      float v = (1.f - zg) * n;
      h2h[m * 64 + j] = (_Float16)v;
      xin[m * 256 + LD + j] = (_Float16)v;  // lastWord for next step
    }
    __syncthreads();

    // fused fc2: pred_t = h2(16x64) @ fc2^T (N=64 -> 4 tiles, one per wave)
    {
      int nt = wave;
      v8f a = {};
#pragma unroll
      for (int kt = 0; kt < 2; ++kt)
        a = wmma16(load_frag_a(h2h + kt * 32, 64),
                   load_frag_b_sw(frag_ptr(fc2t, nt, kt, 2)), a);
      int lane = tid & 31;
      int n = lane & 15, mb = (lane >> 4) << 3;
      int ng = nt * 16 + n;
      float bn = fc2_b[ng];
#pragma unroll
      for (int v = 0; v < 8; ++v) {
        int m = mb + v;
        pred[((size_t)(b0 + m) * TLEN + t) * KN + ng] = a[v] + bn;
      }
    }
    __syncthreads();
  }
}

extern "C" void kernel_launch(void* const* d_in, const int* in_sizes, int n_in,
                              void* d_out, int out_size, void* d_ws, size_t ws_size,
                              hipStream_t stream) {
  const int*   x       = (const int*)d_in[0];
  const float* emb     = (const float*)d_in[1];
  const float* eW_ih0  = (const float*)d_in[2];
  const float* eW_hh0  = (const float*)d_in[3];
  const float* eb_ih0  = (const float*)d_in[4];
  const float* eb_hh0  = (const float*)d_in[5];
  const float* eW_ih1  = (const float*)d_in[6];
  const float* eW_hh1  = (const float*)d_in[7];
  const float* eb_ih1  = (const float*)d_in[8];
  const float* eb_hh1  = (const float*)d_in[9];
  const float* dW_ih0  = (const float*)d_in[10];
  const float* db_ih0  = (const float*)d_in[12];
  const float* db_hh0  = (const float*)d_in[13];
  const float* dW_ih1  = (const float*)d_in[14];
  const float* db_ih1  = (const float*)d_in[16];
  const float* db_hh1  = (const float*)d_in[17];
  const float* fc11_w  = (const float*)d_in[18];
  const float* fc11_b  = (const float*)d_in[19];
  const float* fc12_w  = (const float*)d_in[20];
  const float* fc12_b  = (const float*)d_in[21];
  const float* p1_w    = (const float*)d_in[22];
  const float* p1_b    = (const float*)d_in[23];
  const float* p2_w    = (const float*)d_in[24];
  const float* p2_b    = (const float*)d_in[25];
  const float* fc2_w   = (const float*)d_in[26];
  const float* fc2_b   = (const float*)d_in[27];
  const float* eps     = (const float*)d_in[28];

  // workspace layout (halfs; every region a multiple of 16 halfs = 32B)
  _Float16* wsh = (_Float16*)d_ws;
  size_t o = 0;
  _Float16* Bih0e = wsh + o; o += (size_t)64 * 608;
  _Float16* Bhh0e = wsh + o; o += (size_t)224 * 608;
  _Float16* Bih1e = wsh + o; o += (size_t)224 * 608;
  _Float16* Bhh1e = wsh + o; o += (size_t)224 * 608;
  _Float16* fc11t = wsh + o; o += (size_t)224 * 208;
  _Float16* fc12t = wsh + o; o += (size_t)224 * 208;
  _Float16* p1t   = wsh + o; o += (size_t)224 * 208;
  _Float16* dih0t = wsh + o; o += (size_t)256 * 112;
  _Float16* dih1t = wsh + o; o += (size_t)64 * 112;
  _Float16* fc2t  = wsh + o; o += (size_t)64 * 64;
  _Float16* h2f16 = wsh + o; o += (size_t)BATCH * 224;
  _Float16* zf16  = wsh + o; o += (size_t)BATCH * LD;

  auto cv = [&](const float* W, _Float16* dst, int Nr, int Kd, int Kp, int Np) {
    int total = Kp * Np;
    convert_wT_swz_kernel<<<(total + 255) / 256, 256, 0, stream>>>(W, dst, Nr, Kd, Kp, Np);
  };
  cv(eW_ih0, Bih0e, 3 * LD, ED,      64,  608);
  cv(eW_hh0, Bhh0e, 3 * LD, LD,      224, 608);
  cv(eW_ih1, Bih1e, 3 * LD, LD,      224, 608);
  cv(eW_hh1, Bhh1e, 3 * LD, LD,      224, 608);
  cv(fc11_w, fc11t, LD,     LD,      224, 208);
  cv(fc12_w, fc12t, LD,     LD,      224, 208);
  cv(p1_w,   p1t,   LD,     LD,      224, 208);
  cv(dW_ih0, dih0t, 3 * ED, LD + ED, 256, 112);
  cv(dW_ih1, dih1t, 3 * ED, ED,      64,  112);
  cv(fc2_w,  fc2t,  KN,     ED,      64,  64);

  float* out    = (float*)d_out;
  float* out_mu = out + (size_t)BATCH * TLEN * KN;
  float* out_lv = out_mu + (size_t)BATCH * LD;
  float* out_py = out_lv + (size_t)BATCH * LD;

  encoder_kernel<<<BATCH / 16, 256, 0, stream>>>(
      x, emb, Bih0e, Bhh0e, Bih1e, Bhh1e, eb_ih0, eb_hh0, eb_ih1, eb_hh1, h2f16);

  latent_kernel<<<BATCH / 16, 256, 0, stream>>>(
      h2f16, fc11t, fc12t, p1t, fc11_b, fc12_b, p1_b, p2_w, p2_b, eps,
      out_mu, out_lv, out_py, zf16);

  decoder_kernel<<<BATCH / 16, 128, 0, stream>>>(
      zf16, dih0t, dih1t, fc2t, db_ih0, db_hh0, db_ih1, db_hh1, fc2_b, out);
}
